// GaussianTanh_40192303956301
// MI455X (gfx1250) — matrix operand order repaired and checked by reading, not compiled, not run on hardware
//
#include <hip/hip_runtime.h>
#include <cstdint>

// ---------------------------------------------------------------------------
// Fused conditional-Gaussian-tanh sampler for MI455X (gfx1250, wave32, WMMA).
//   - 3 MLPs (256->256->256->{16,16,120}) run on v_wmma_f32_16x16x32_f16
//   - weights pre-transposed/f16-converted into d_ws by a prep kernel
//   - weight tiles staged into LDS with the Tensor Data Mover
//   - per-row MVN/tanh math done in registers at the end of the block
// ---------------------------------------------------------------------------

typedef _Float16 v16h __attribute__((ext_vector_type(16)));
typedef _Float16 v8h  __attribute__((ext_vector_type(8)));
typedef float    v8f  __attribute__((ext_vector_type(8)));
typedef unsigned int  u32x4 __attribute__((ext_vector_type(4)));
typedef int           i32x4 __attribute__((ext_vector_type(4)));
typedef int           i32x8 __attribute__((ext_vector_type(8)));

union H16 { v16h v; v8h h[2]; uint4 u[2]; };

#define RS       264                       // halves per row (256 + 8 pad -> 528B rows, bank-staggered)
#define ROWS_PB  128                       // rows per block (8 waves x 16 rows)
#define WS_BYTES 67584                     // 128 * 264 * 2  (one 128-row weight chunk)
#define HA_OFF   67584
#define HB_OFF   135168
#define MU_OFF   202752                    // 128*17*4 = 8704
#define LA_OFF   211456                    // 128*17*4 = 8704
#define OF_OFF   220160                    // 128*121*4 = 61952
#define SMEM_BYTES 282112

// ws (half-element) offsets of the 9 transposed weight matrices
#define O_MU_W1  0u
#define O_MU_W2  67584u
#define O_MU_W3  135168u
#define O_LA_W1  139392u
#define O_LA_W2  206976u
#define O_LA_W3  274560u
#define O_OF_W1  278784u
#define O_OF_W2  346368u
#define O_OF_W3  413952u
#define WS_TOTAL_HALVES 447744u

// ---------------------------------------------------------------------------
// TDM staging: flat copy of `nbytes` (4B elements, 1 row tile) global -> LDS.
// Descriptor per CDNA5 ISA ch.8 (group0: count/lds/global/type, group1:
// data_size=4B, tensor_dim0 = tile_dim0 = ndwords, tile_dim1 = 1).
// ---------------------------------------------------------------------------
__device__ inline void tdm_stage(void* ldsDst, const void* gsrc, unsigned nbytes)
{
  unsigned ndw     = nbytes >> 2;                      // <= 16896, fits tile_dim0
  unsigned lds_off = (unsigned)(uintptr_t)ldsDst;      // LDS aperture: low 32 bits = offset
  unsigned long long ga = (unsigned long long)(uintptr_t)gsrc;

  u32x4 g0 = { 1u,                                     // count=1 (valid user descriptor)
               lds_off,                                 // lds_addr
               (unsigned)(ga & 0xFFFFFFFFu),            // global_addr[31:0]
               (unsigned)((ga >> 32) & 0x1FFFFFFu) | (2u << 30) }; // addr[56:32] | type=2

  i32x8 g1 = { (int)0x00020000u,                        // workgroup_mask=0, data_size=2 (4B)
               (int)((ndw & 0xFFFFu) << 16),            // tensor_dim0[15:0]
               (int)(((ndw >> 16) & 0xFFFFu) | (1u << 16)), // tensor_dim0[31:16] | tensor_dim1=1
               (int)((ndw & 0xFFFFu) << 16),            // tensor_dim1 hi=0 | tile_dim0=ndw
               1,                                        // tile_dim1=1, tile_dim2=0
               (int)ndw,                                 // tensor_dim0_stride[31:0]
               0, 0 };
  i32x4 gz = { 0, 0, 0, 0 };
#if __has_include(<hip/amd_detail/amd_gfx1250_TDM.h>)
  i32x8 gz8 = { 0, 0, 0, 0, 0, 0, 0, 0 };
  __builtin_amdgcn_tensor_load_to_lds(g0, g1, gz, gz, gz8, 0);
#else
  __builtin_amdgcn_tensor_load_to_lds(g0, g1, gz, gz, 0);
#endif
  __builtin_amdgcn_s_wait_tensorcnt(0);
}

// ---------------------------------------------------------------------------
// Fragment helpers (CDNA5 16-bit WMMA layouts, wave32):
//   A 16x32:  lane(0..15)=row M, halves[0..7]=K(k0+hi*8+i), halves[8..15]=+16
//   B 32x16:  lane&15 = col N, halves[0..7]=K(k0+hi*8+i), halves[8..15]=+16 (as A)
//   C/D f32:  lane&15 = N, vgpr r -> M = r + 8*(lane>=16)
// ---------------------------------------------------------------------------
__device__ inline v8h cvt8(float4 a, float4 b)
{
  v8h r;
  r[0] = (_Float16)a.x; r[1] = (_Float16)a.y; r[2] = (_Float16)a.z; r[3] = (_Float16)a.w;
  r[4] = (_Float16)b.x; r[5] = (_Float16)b.y; r[6] = (_Float16)b.z; r[7] = (_Float16)b.w;
  return r;
}

__device__ inline void load_a_global(H16 (&af)[8], const float* __restrict__ rowp, int hi)
{
#pragma unroll
  for (int ks = 0; ks < 8; ++ks) {
    int kb = ks * 32 + hi * 8;
    float4 a0 = *(const float4*)(rowp + kb);
    float4 a1 = *(const float4*)(rowp + kb + 4);
    float4 a2 = *(const float4*)(rowp + kb + 16);
    float4 a3 = *(const float4*)(rowp + kb + 20);
    af[ks].h[0] = cvt8(a0, a1);
    af[ks].h[1] = cvt8(a2, a3);
  }
}

__device__ inline void load_a_lds(H16 (&af)[8], const _Float16* hrow, int hi)
{
#pragma unroll
  for (int ks = 0; ks < 8; ++ks) {
    int kb = ks * 32 + hi * 8;
    af[ks].u[0] = *(const uint4*)(hrow + kb);
    af[ks].u[1] = *(const uint4*)(hrow + kb + 16);
  }
}

// One 256->256 hidden layer with relu: stage W^T in two 128-col chunks.
__device__ inline void hidden_layer(const H16 (&af)[8], const _Float16* __restrict__ wsrc,
                                    _Float16* ldsW, _Float16* hOut,
                                    const float* __restrict__ bias, int wave, int lane)
{
  const int m = lane & 15, hi = lane >> 4;
  for (int half = 0; half < 2; ++half) {
    __syncthreads();                               // all waves done reading previous Wstage
    if (wave == 0) tdm_stage(ldsW, wsrc + (unsigned)half * (128u * RS), WS_BYTES);
    __syncthreads();
    for (int nt = 0; nt < 8; ++nt) {
      v8f acc = { 0.f, 0.f, 0.f, 0.f, 0.f, 0.f, 0.f, 0.f };
#pragma unroll
      for (int ks = 0; ks < 8; ++ks) {
        H16 bf;
        const _Float16* wp = ldsW + (nt * 16 + m) * RS + ks * 32 + hi * 8;
        bf.u[0] = *(const uint4*)wp;
        bf.u[1] = *(const uint4*)(wp + 16);
        acc = __builtin_amdgcn_wmma_f32_16x16x32_f16(false, af[ks].v, false, bf.v,
                                                     (short)0, acc, false, false);
      }
      const int n = half * 128 + nt * 16 + m;
      const float bv = bias[n];
#pragma unroll
      for (int r = 0; r < 8; ++r) {
        float x = fmaxf(acc[r] + bv, 0.f);
        hOut[(wave * 16 + r + 8 * hi) * RS + n] = (_Float16)x;
      }
    }
  }
}

// Final 256->Nout layer, results to f32 LDS tile (no activation).
__device__ inline void out_layer(const H16 (&af)[8], const _Float16* __restrict__ wsrc,
                                 _Float16* ldsW, float* outBuf, int ostride,
                                 const float* __restrict__ bias, int Nout, int Npad,
                                 int wave, int lane)
{
  const int m = lane & 15, hi = lane >> 4;
  __syncthreads();
  if (wave == 0) tdm_stage(ldsW, wsrc, (unsigned)Npad * RS * 2u);
  __syncthreads();
  const int ntiles = Npad / 16;
  for (int nt = 0; nt < ntiles; ++nt) {
    v8f acc = { 0.f, 0.f, 0.f, 0.f, 0.f, 0.f, 0.f, 0.f };
#pragma unroll
    for (int ks = 0; ks < 8; ++ks) {
      H16 bf;
      const _Float16* wp = ldsW + (nt * 16 + m) * RS + ks * 32 + hi * 8;
      bf.u[0] = *(const uint4*)wp;
      bf.u[1] = *(const uint4*)(wp + 16);
      acc = __builtin_amdgcn_wmma_f32_16x16x32_f16(false, af[ks].v, false, bf.v,
                                                   (short)0, acc, false, false);
    }
    const int n = nt * 16 + m;
    if (n < Nout) {
      const float bv = bias[n];
#pragma unroll
      for (int r = 0; r < 8; ++r)
        outBuf[(wave * 16 + r + 8 * hi) * ostride + n] = acc[r] + bv;
    }
  }
}

// ---------------------------------------------------------------------------
// Prep: W (K=256 x N, f32 row-major) -> W^T f16 [Npad][RS], zero-padded.
// ---------------------------------------------------------------------------
__global__ void prep_wt_kernel(const float* __restrict__ W, _Float16* __restrict__ dst,
                               int N, int Npad)
{
  int idx = blockIdx.x * 256 + threadIdx.x;
  int total = Npad * RS;
  if (idx >= total) return;
  int n = idx / RS;
  int k = idx - n * RS;
  float v = (k < 256 && n < N) ? W[k * N + n] : 0.f;
  dst[idx] = (_Float16)v;
}

// ---------------------------------------------------------------------------
// Main fused kernel: one block = 128 rows.
// ---------------------------------------------------------------------------
__global__ __launch_bounds__(256, 1)
void fused_gauss_tanh_kernel(const float* __restrict__ cond, const float* __restrict__ epsg,
                             const float* b_mu1, const float* b_mu2, const float* b_mu3,
                             const float* b_la1, const float* b_la2, const float* b_la3,
                             const float* b_of1, const float* b_of2, const float* b_of3,
                             const _Float16* __restrict__ wsW,
                             float* __restrict__ out, int Btot)
{
  extern __shared__ char smem[];
  _Float16* ldsW = (_Float16*)(smem);
  _Float16* hA   = (_Float16*)(smem + HA_OFF);
  _Float16* hB   = (_Float16*)(smem + HB_OFF);
  float*    muB  = (float*)(smem + MU_OFF);
  float*    laB  = (float*)(smem + LA_OFF);
  float*    ofB  = (float*)(smem + OF_OFF);

  const int tid  = threadIdx.x;
  const int wave = tid >> 5;
  const int lane = tid & 31;
  const int m    = lane & 15;
  const int hi   = lane >> 4;
  const int row0 = blockIdx.x * ROWS_PB;
  const int rowInTile = wave * 16 + m;
  const size_t grow = (size_t)row0 + rowInTile;

  // condition A-fragments: loaded once, reused by all three MLPs' layer 1
  H16 afC[8];
  load_a_global(afC, cond + grow * 256, hi);

  const float* B1[3] = { b_mu1, b_la1, b_of1 };
  const float* B2[3] = { b_mu2, b_la2, b_of2 };
  const float* B3[3] = { b_mu3, b_la3, b_of3 };
  const unsigned W1O[3] = { O_MU_W1, O_LA_W1, O_OF_W1 };
  const unsigned W2O[3] = { O_MU_W2, O_LA_W2, O_OF_W2 };
  const unsigned W3O[3] = { O_MU_W3, O_LA_W3, O_OF_W3 };
  float* OUT3[3] = { muB, laB, ofB };
  const int OST[3]  = { 17, 17, 121 };
  const int NOUT[3] = { 16, 16, 120 };
  const int NPAD[3] = { 16, 16, 128 };

#pragma unroll
  for (int mlp = 0; mlp < 3; ++mlp) {
    hidden_layer(afC, wsW + W1O[mlp], ldsW, hA, B1[mlp], wave, lane);
    H16 af1[8];
    load_a_lds(af1, hA + rowInTile * RS, hi);     // wave-local rows: no barrier needed
    hidden_layer(af1, wsW + W2O[mlp], ldsW, hB, B2[mlp], wave, lane);
    H16 af2[8];
    load_a_lds(af2, hB + rowInTile * RS, hi);
    out_layer(af2, wsW + W3O[mlp], ldsW, OUT3[mlp], OST[mlp], B3[mlp],
              NOUT[mlp], NPAD[mlp], wave, lane);
  }

  __syncthreads();

  // ---------------- per-row MVN + tanh epilogue (128 threads, 1 row each) ---
  if (tid < ROWS_PB) {
    const int r = tid;
    const size_t gr = (size_t)row0 + r;
    const float* mup = muB + r * 17;
    const float* lap = laB + r * 17;
    float* ofp = ofB + r * 121;

#pragma unroll
    for (int i = 0; i < 120; ++i) ofp[i] = tanhf(ofp[i]);   // off = tanh(raw)

    float lam[16], epsv[16];
#pragma unroll
    for (int j = 0; j < 16; ++j) lam[j] = 1e-8f + fmaxf(lap[j], 0.f);
    const float* ep = epsg + gr * 16;
#pragma unroll
    for (int j = 0; j < 16; ++j) epsv[j] = ep[j];

    // t = L*eps  (L[j][k] = lam[j]*lam[k]*off, diag = lam^2)
    float t[16];
#pragma unroll
    for (int j = 0; j < 16; ++j) {
      float s = lam[j] * lam[j] * epsv[j];
#pragma unroll
      for (int k = 0; k < j; ++k)
        s += lam[j] * lam[k] * ofp[j * (j - 1) / 2 + k] * epsv[k];
      t[j] = s;
    }

    // forward solve L z = t; accumulate log-prob pieces
    float z[16], sumz2 = 0.f, sumld = 0.f;
#pragma unroll
    for (int j = 0; j < 16; ++j) {
      float s = t[j];
#pragma unroll
      for (int k = 0; k < j; ++k)
        s -= lam[j] * lam[k] * ofp[j * (j - 1) / 2 + k] * z[k];
      const float dj = lam[j] * lam[j];
      z[j] = s / dj;
      sumz2 += z[j] * z[j];
      sumld += logf(dj);
    }
    const float base_lp = -0.5f * sumz2 - 16.f * 0.91893853320467274f - sumld;

    float ladj = 0.f;
    float yv[16];
#pragma unroll
    for (int j = 0; j < 16; ++j) {
      const float x0 = mup[j] + t[j];
      yv[j] = tanhf(x0);
      const float v = -2.f * x0;
      const float sp = fmaxf(v, 0.f) + log1pf(expf(-fabsf(v)));  // softplus(-2x)
      ladj += 2.f * (0.69314718055994531f - x0 - sp);
    }

    float* yout = out + gr * 16;
#pragma unroll
    for (int j = 0; j < 16; ++j) yout[j] = yv[j];
    out[(size_t)Btot * 16 + gr] = base_lp - ladj;
  }
}

// ---------------------------------------------------------------------------
extern "C" void kernel_launch(void* const* d_in, const int* in_sizes, int n_in,
                              void* d_out, int out_size, void* d_ws, size_t ws_size,
                              hipStream_t stream)
{
  const float* cond = (const float*)d_in[0];
  const float* eps  = (const float*)d_in[1];
  _Float16* wsh = (_Float16*)d_ws;
  const int Btot = in_sizes[0] / 256;

  // ---- prep: transpose + f16-convert the 9 weight matrices into ws --------
  struct Mat { const float* w; unsigned off; int N; int Npad; };
  const Mat mats[9] = {
    { (const float*)d_in[2],  O_MU_W1, 256, 256 },
    { (const float*)d_in[4],  O_MU_W2, 256, 256 },
    { (const float*)d_in[6],  O_MU_W3,  16,  16 },
    { (const float*)d_in[8],  O_LA_W1, 256, 256 },
    { (const float*)d_in[10], O_LA_W2, 256, 256 },
    { (const float*)d_in[12], O_LA_W3,  16,  16 },
    { (const float*)d_in[14], O_OF_W1, 256, 256 },
    { (const float*)d_in[16], O_OF_W2, 256, 256 },
    { (const float*)d_in[18], O_OF_W3, 120, 128 },
  };
  for (int i = 0; i < 9; ++i) {
    const int total = mats[i].Npad * RS;
    prep_wt_kernel<<<(total + 255) / 256, 256, 0, stream>>>(
        mats[i].w, wsh + mats[i].off, mats[i].N, mats[i].Npad);
  }

  // ---- main fused kernel ---------------------------------------------------
  (void)hipFuncSetAttribute((const void*)fused_gauss_tanh_kernel,
                            hipFuncAttributeMaxDynamicSharedMemorySize, SMEM_BYTES);
  fused_gauss_tanh_kernel<<<Btot / ROWS_PB, 256, SMEM_BYTES, stream>>>(
      cond, eps,
      (const float*)d_in[3],  (const float*)d_in[5],  (const float*)d_in[7],
      (const float*)d_in[9],  (const float*)d_in[11], (const float*)d_in[13],
      (const float*)d_in[15], (const float*)d_in[17], (const float*)d_in[19],
      wsh, (float*)d_out, Btot);
}
